// MultiheadAttn_14190571946297
// MI455X (gfx1250) — matrix operand order repaired and checked by reading, not compiled, not run on hardware
//
#include <hip/hip_runtime.h>

// ---------------- problem constants ----------------
#define TT 2048
#define BB 2
#define EE 1024
#define HH 16
#define HD 64
#define F3 (3 * EE)          // 3072
#define MROWS (TT * BB)      // 4096
#define SSTR 2064            // padded LDS score row stride (f32)
#define ASTR 1032            // padded LDS A-tile row stride (bf16 elems)

typedef __attribute__((ext_vector_type(16))) __bf16          v16bf;
typedef __attribute__((ext_vector_type(8)))  float           v8f;
typedef __attribute__((ext_vector_type(8)))  unsigned short  u16x8;

struct U16Pair { u16x8 lo, hi; };   // 32 bytes == v16bf

__device__ __forceinline__ unsigned short f2bf(float f) {
  unsigned int u = __builtin_bit_cast(unsigned int, f);
  u += 0x7FFFu + ((u >> 16) & 1u);         // round-to-nearest-even
  return (unsigned short)(u >> 16);
}

// A-matrix (16x32 bf16) per-lane fragment: row = lane&15,
// elems 0..7  -> K = k0 + 2*klo + j      (klo = 8*(lane>=16))
// elems 8..15 -> K = k0 + 2*klo + 8 + j
__device__ __forceinline__ v16bf make_a(const unsigned short* row, int k0, int klo) {
  U16Pair p;
  p.lo = *(const u16x8*)(row + k0 + 2 * klo);
  p.hi = *(const u16x8*)(row + k0 + 2 * klo + 8);
  return __builtin_bit_cast(v16bf, p);
}

// ---- CDNA5 async global->LDS copy (ASYNCcnt-tracked, ISA 08 §4) ----
__device__ __forceinline__ void async_g2l_b128(unsigned lds_off,
                                               const unsigned short* g) {
  asm volatile("global_load_async_to_lds_b128 %0, %1, off"
               :: "v"(lds_off), "v"(g) : "memory");
}
__device__ __forceinline__ void wait_async0() {
  asm volatile("s_wait_asynccnt 0" ::: "memory");
}

// ---------------- helpers: convert / zero ----------------
__global__ void cvt_bf16(const float* __restrict__ s, unsigned short* __restrict__ d, int n) {
  int i = blockIdx.x * blockDim.x + threadIdx.x;
  int st = gridDim.x * blockDim.x;
  for (; i < n; i += st) d[i] = f2bf(s[i]);
}

__global__ void zero_f32(float* __restrict__ p, int n) {
  int i = blockIdx.x * blockDim.x + threadIdx.x;
  int st = gridDim.x * blockDim.x;
  for (; i < n; i += st) p[i] = 0.0f;
}

// ---------------- QKV projection ----------------
// C[M=4096, N=3072] = Xbf[M,1024] * Win[N,1024]^T + bias ; q scaled by 1/8.
// WG: 32 M-rows staged in LDS (async); wave owns 32x64 (2 M-tiles x 4 N-tiles).
// q,k stored head-major [B*H][T][HD]; v stored TRANSPOSED per head [B*H][HD][T].
__global__ __launch_bounds__(256)
void qkv_gemm(const unsigned short* __restrict__ xbf,
              const unsigned short* __restrict__ wbf,
              const float* __restrict__ bias,
              unsigned short* __restrict__ qh,
              unsigned short* __restrict__ kh,
              unsigned short* __restrict__ vt) {
  extern __shared__ unsigned short sA[];     // 32 x ASTR bf16 = ~66 KB

  const int wave = threadIdx.x >> 5;
  const int lane = threadIdx.x & 31;
  const int hi   = lane >> 4;
  const int ln   = lane & 15;
  const int klo  = hi * 8;
  const int kb   = hi * 16;
  const int row0 = blockIdx.y * 32;
  const int col0 = blockIdx.x * 512 + wave * 64;   // 64-aligned -> one head, one q/k/v

  // stage A tile (32 rows x 1024 bf16) into LDS with async b128 copies
  {
    int c = threadIdx.x;               // 4096 16-byte chunks / 256 threads
    #pragma unroll
    for (int it = 0; it < 16; ++it, c += 256) {
      const int row = c >> 7, col = c & 127;
      const unsigned loff = (unsigned)(size_t)&sA[row * ASTR + col * 8];
      async_g2l_b128(loff, xbf + (size_t)(row0 + row) * EE + col * 8);
    }
    wait_async0();
    __syncthreads();
  }

  const unsigned short* arow0 = &sA[ln * ASTR];
  const unsigned short* arow1 = &sA[(16 + ln) * ASTR];
  const unsigned short* brow[4];
  #pragma unroll
  for (int j = 0; j < 4; ++j)
    brow[j] = wbf + (size_t)(col0 + j * 16 + ln) * EE;   // B[k][n] = W[n][k]

  v8f acc[2][4] = {};
  #pragma unroll 2
  for (int k0 = 0; k0 < EE; k0 += 32) {
    v16bf a0 = make_a(arow0, k0, klo);
    v16bf a1 = make_a(arow1, k0, klo);
    v16bf b0 = *(const v16bf*)(brow[0] + k0 + kb);
    v16bf b1 = *(const v16bf*)(brow[1] + k0 + kb);
    v16bf b2 = *(const v16bf*)(brow[2] + k0 + kb);
    v16bf b3 = *(const v16bf*)(brow[3] + k0 + kb);
    acc[0][0] = __builtin_amdgcn_wmma_f32_16x16x32_bf16(false, a0, false, b0, (short)0, acc[0][0], false, false);
    acc[0][1] = __builtin_amdgcn_wmma_f32_16x16x32_bf16(false, a0, false, b1, (short)0, acc[0][1], false, false);
    acc[0][2] = __builtin_amdgcn_wmma_f32_16x16x32_bf16(false, a0, false, b2, (short)0, acc[0][2], false, false);
    acc[0][3] = __builtin_amdgcn_wmma_f32_16x16x32_bf16(false, a0, false, b3, (short)0, acc[0][3], false, false);
    acc[1][0] = __builtin_amdgcn_wmma_f32_16x16x32_bf16(false, a1, false, b0, (short)0, acc[1][0], false, false);
    acc[1][1] = __builtin_amdgcn_wmma_f32_16x16x32_bf16(false, a1, false, b1, (short)0, acc[1][1], false, false);
    acc[1][2] = __builtin_amdgcn_wmma_f32_16x16x32_bf16(false, a1, false, b2, (short)0, acc[1][2], false, false);
    acc[1][3] = __builtin_amdgcn_wmma_f32_16x16x32_bf16(false, a1, false, b3, (short)0, acc[1][3], false, false);
  }

  // 64-col block lies inside one of q/k/v and one head
  const int   sel = col0 / EE;          // 0=q 1=k 2=v
  const int   fl0 = col0 % EE;
  const int   h   = fl0 / HD;
  const float scl = (sel == 0) ? 0.125f : 1.0f;
  #pragma unroll
  for (int j = 0; j < 4; ++j) {
    const int   d  = (fl0 % HD) + j * 16 + ln;   // feature within head (fl0%HD==0)
    const float bv = bias[col0 + j * 16 + ln];
    #pragma unroll
    for (int m2 = 0; m2 < 2; ++m2) {
      #pragma unroll
      for (int r = 0; r < 8; ++r) {
        const int mg   = row0 + m2 * 16 + r + hi * 8;
        const int t    = mg / BB;
        const int bidx = mg % BB;
        const int head = bidx * HH + h;
        const unsigned short val = f2bf((acc[m2][j][r] + bv) * scl);
        if (sel == 0)      qh[((size_t)head * TT + t) * HD + d] = val;
        else if (sel == 1) kh[((size_t)head * TT + t) * HD + d] = val;
        else               vt[((size_t)head * HD + d) * TT + t] = val;  // transposed
      }
    }
  }
}

// ---------------- attention core ----------------
__global__ __launch_bounds__(256)
void attn_kernel(const unsigned short* __restrict__ qh,
                 const unsigned short* __restrict__ kh,
                 const unsigned short* __restrict__ vt,
                 unsigned short* __restrict__ attnx,
                 float* __restrict__ avg) {
  extern __shared__ float smem[];
  float* sc  = smem;                 // [16][SSTR]
  float* red = smem + 16 * SSTR;     // [16][16]

  const int qt   = blockIdx.x;
  const int head = blockIdx.y;
  const int bidx = head >> 4;
  const int h    = head & 15;
  const int wave = threadIdx.x >> 5;
  const int lane = threadIdx.x & 31;
  const int hi   = lane >> 4;
  const int ln   = lane & 15;
  const int klo  = hi * 8;
  const int kb   = hi * 16;

  // --- phase 1: S = q @ k^T into LDS ---
  const unsigned short* qrow = qh + ((size_t)head * TT + qt * 16 + ln) * HD;
  const v16bf a0 = make_a(qrow, 0, klo);
  const v16bf a1 = make_a(qrow, 32, klo);
  for (int nt = wave; nt < TT / 16; nt += 8) {
    const unsigned short* krow = kh + ((size_t)head * TT + nt * 16 + ln) * HD;
    v16bf b0 = *(const v16bf*)(krow + kb);
    v16bf b1 = *(const v16bf*)(krow + 32 + kb);
    v8f acc = {};
    acc = __builtin_amdgcn_wmma_f32_16x16x32_bf16(false, a0, false, b0, (short)0, acc, false, false);
    acc = __builtin_amdgcn_wmma_f32_16x16x32_bf16(false, a1, false, b1, (short)0, acc, false, false);
    #pragma unroll
    for (int r = 0; r < 8; ++r)
      sc[(r + hi * 8) * SSTR + nt * 16 + ln] = acc[r];
  }
  __syncthreads();

  // --- phase 2: softmax per row + atomic head-average ---
  {
    const int row = threadIdx.x >> 4;
    const int seg = threadIdx.x & 15;
    float* p = sc + row * SSTR + seg * 128;
    float mx = -1e30f;
    for (int i = 0; i < 128; ++i) mx = fmaxf(mx, p[i]);
    red[row * 16 + seg] = mx;
    __syncthreads();
    float rmx = -1e30f;
    #pragma unroll
    for (int i = 0; i < 16; ++i) rmx = fmaxf(rmx, red[row * 16 + i]);
    __syncthreads();
    float sum = 0.0f;
    for (int i = 0; i < 128; ++i) { float e = __expf(p[i] - rmx); p[i] = e; sum += e; }
    red[row * 16 + seg] = sum;
    __syncthreads();
    float tot = 0.0f;
    #pragma unroll
    for (int i = 0; i < 16; ++i) tot += red[row * 16 + i];
    const float inv = 1.0f / tot;
    const int tq = qt * 16 + row;
    float* arow_ = avg + ((size_t)bidx * TT + tq) * TT + seg * 128;
    for (int i = 0; i < 128; ++i) {
      float pr = p[i] * inv;
      p[i] = pr;
      atomicAdd(arow_ + i, pr * (1.0f / HH));
    }
  }
  __syncthreads();

  // --- phase 3: O = P @ V (waves 0..3 own four 16-wide d tiles) ---
  // V is transposed per head: vt[head][d][key] -> contiguous B fragments.
  if (wave < 4) {
    const int dt = wave;
    const unsigned short* vrow = vt + ((size_t)head * HD + dt * 16 + ln) * TT;
    v8f acc = {};
    for (int key0 = 0; key0 < TT; key0 += 32) {
      const float* prow = sc + ln * SSTR + key0;
      U16Pair ap;
      #pragma unroll
      for (int j = 0; j < 8; ++j) {
        ap.lo[j] = f2bf(prow[kb + j]);
        ap.hi[j] = f2bf(prow[kb + 8 + j]);
      }
      v16bf a = __builtin_bit_cast(v16bf, ap);
      v16bf b = *(const v16bf*)(vrow + key0 + kb);   // B[k][n] = V[key][d]
      acc = __builtin_amdgcn_wmma_f32_16x16x32_bf16(false, a, false, b, (short)0, acc, false, false);
    }
    #pragma unroll
    for (int r = 0; r < 8; ++r) {
      const int t = qt * 16 + r + hi * 8;
      attnx[((size_t)t * BB + bidx) * EE + h * HD + dt * 16 + ln] = f2bf(acc[r]);
    }
  }
}

// ---------------- output projection ----------------
// out[M=4096, N=1024] = attnx[M,1024] * Wout[N,1024]^T + bias  (fp32 out)
__global__ __launch_bounds__(256)
void out_gemm(const unsigned short* __restrict__ abf,
              const unsigned short* __restrict__ wbf,
              const float* __restrict__ bias,
              float* __restrict__ out) {
  extern __shared__ unsigned short sB[];     // 32 x ASTR bf16

  const int wave = threadIdx.x >> 5;
  const int lane = threadIdx.x & 31;
  const int hi   = lane >> 4;
  const int ln   = lane & 15;
  const int klo  = hi * 8;
  const int kb   = hi * 16;
  const int row0 = blockIdx.y * 32;
  const int col0 = blockIdx.x * 512 + wave * 64;

  {
    int c = threadIdx.x;
    #pragma unroll
    for (int it = 0; it < 16; ++it, c += 256) {
      const int row = c >> 7, col = c & 127;
      const unsigned loff = (unsigned)(size_t)&sB[row * ASTR + col * 8];
      async_g2l_b128(loff, abf + (size_t)(row0 + row) * EE + col * 8);
    }
    wait_async0();
    __syncthreads();
  }

  const unsigned short* arow0 = &sB[ln * ASTR];
  const unsigned short* arow1 = &sB[(16 + ln) * ASTR];
  const unsigned short* brow[4];
  #pragma unroll
  for (int j = 0; j < 4; ++j)
    brow[j] = wbf + (size_t)(col0 + j * 16 + ln) * EE;

  v8f acc[2][4] = {};
  #pragma unroll 2
  for (int k0 = 0; k0 < EE; k0 += 32) {
    v16bf a0 = make_a(arow0, k0, klo);
    v16bf a1 = make_a(arow1, k0, klo);
    v16bf b0 = *(const v16bf*)(brow[0] + k0 + kb);
    v16bf b1 = *(const v16bf*)(brow[1] + k0 + kb);
    v16bf b2 = *(const v16bf*)(brow[2] + k0 + kb);
    v16bf b3 = *(const v16bf*)(brow[3] + k0 + kb);
    acc[0][0] = __builtin_amdgcn_wmma_f32_16x16x32_bf16(false, a0, false, b0, (short)0, acc[0][0], false, false);
    acc[0][1] = __builtin_amdgcn_wmma_f32_16x16x32_bf16(false, a0, false, b1, (short)0, acc[0][1], false, false);
    acc[0][2] = __builtin_amdgcn_wmma_f32_16x16x32_bf16(false, a0, false, b2, (short)0, acc[0][2], false, false);
    acc[0][3] = __builtin_amdgcn_wmma_f32_16x16x32_bf16(false, a0, false, b3, (short)0, acc[0][3], false, false);
    acc[1][0] = __builtin_amdgcn_wmma_f32_16x16x32_bf16(false, a1, false, b0, (short)0, acc[1][0], false, false);
    acc[1][1] = __builtin_amdgcn_wmma_f32_16x16x32_bf16(false, a1, false, b1, (short)0, acc[1][1], false, false);
    acc[1][2] = __builtin_amdgcn_wmma_f32_16x16x32_bf16(false, a1, false, b2, (short)0, acc[1][2], false, false);
    acc[1][3] = __builtin_amdgcn_wmma_f32_16x16x32_bf16(false, a1, false, b3, (short)0, acc[1][3], false, false);
  }

  #pragma unroll
  for (int j = 0; j < 4; ++j) {
    const float bv = bias[col0 + j * 16 + ln];
    #pragma unroll
    for (int m2 = 0; m2 < 2; ++m2) {
      #pragma unroll
      for (int r = 0; r < 8; ++r) {
        const int mg = row0 + m2 * 16 + r + hi * 8;
        out[(size_t)mg * EE + col0 + j * 16 + ln] = acc[m2][j][r] + bv;
      }
    }
  }
}

// ---------------- host launcher ----------------
extern "C" void kernel_launch(void* const* d_in, const int* in_sizes, int n_in,
                              void* d_out, int out_size, void* d_ws, size_t ws_size,
                              hipStream_t stream) {
  (void)in_sizes; (void)n_in; (void)out_size; (void)ws_size;
  const float* x     = (const float*)d_in[0];
  const float* in_w  = (const float*)d_in[1];
  const float* in_b  = (const float*)d_in[2];
  const float* out_w = (const float*)d_in[3];
  const float* out_b = (const float*)d_in[4];

  float* out = (float*)d_out;                        // [T,B,E]
  float* avg = out + (size_t)TT * BB * EE;           // [B,T,T]

  unsigned short* xbf    = (unsigned short*)d_ws;
  unsigned short* wInBf  = xbf    + (size_t)MROWS * EE;
  unsigned short* wOutBf = wInBf  + (size_t)F3 * EE;
  unsigned short* qh     = wOutBf + (size_t)EE * EE;
  unsigned short* kh     = qh     + (size_t)BB * HH * TT * HD;
  unsigned short* vt     = kh     + (size_t)BB * HH * TT * HD;  // [B*H][HD][T]
  unsigned short* attnx  = vt     + (size_t)BB * HH * TT * HD;

  cvt_bf16<<<1024, 256, 0, stream>>>(x, xbf, MROWS * EE);
  cvt_bf16<<<1024, 256, 0, stream>>>(in_w, wInBf, F3 * EE);
  cvt_bf16<<<512, 256, 0, stream>>>(out_w, wOutBf, EE * EE);
  zero_f32<<<4096, 256, 0, stream>>>(avg, BB * TT * TT);

  const int gemm_smem = 32 * ASTR * (int)sizeof(unsigned short);  // ~66 KB
  hipFuncSetAttribute((const void*)qkv_gemm,
                      hipFuncAttributeMaxDynamicSharedMemorySize, gemm_smem);
  hipFuncSetAttribute((const void*)out_gemm,
                      hipFuncAttributeMaxDynamicSharedMemorySize, gemm_smem);

  qkv_gemm<<<dim3(F3 / 512, MROWS / 32), 256, gemm_smem, stream>>>(
      xbf, wInBf, in_b, qh, kh, vt);

  const size_t attn_smem = (size_t)(16 * SSTR + 256) * sizeof(float);  // ~130 KB
  hipFuncSetAttribute((const void*)attn_kernel,
                      hipFuncAttributeMaxDynamicSharedMemorySize, (int)attn_smem);
  attn_kernel<<<dim3(TT / 16, BB * HH), 256, attn_smem, stream>>>(
      qh, kh, vt, attnx, avg);

  out_gemm<<<dim3(EE / 512, MROWS / 32), 256, gemm_smem, stream>>>(
      attnx, wOutBf, out_b, out);
}
